// RelativeAttentionHead_23871428231899
// MI455X (gfx1250) — compile-verified, matrix operand order checked by reading, not run on hardware
//
#include <hip/hip_runtime.h>
#include <hip/hip_bf16.h>

// Problem constants (match reference)
#define B_    16
#define T_    1024
#define C_    768
#define HS_   64
#define NREL  2047
#define EPAD  2112   // >= 2047+48 window overreach, multiple of 16, zero padded

typedef __bf16 bf16_t;
typedef __bf16 v16bf __attribute__((ext_vector_type(16)));
typedef __bf16 v8bf  __attribute__((ext_vector_type(8)));
typedef float  v8f   __attribute__((ext_vector_type(8)));

union AF { v16bf v; v8bf h[2]; };

// Load a 16x32 bf16 WMMA fragment (A layout; B layout is the mirrored
// equivalent with 'row' meaning the N column of the pre-transposed operand).
// Per ISA: lanes 0-15 hold K = k0+[0..7] and k0+[16..23]; lanes 16-31 hold
// K = k0+[8..15] and k0+[24..31]; two contiguous 16B chunks per lane.
__device__ __forceinline__ v16bf frag_ld(const bf16_t* rowp, int k0, int half) {
  AF f;
  f.h[0] = *(const v8bf*)(rowp + k0 + half * 8);
  f.h[1] = *(const v8bf*)(rowp + k0 + 16 + half * 8);
  return f.v;
}

__device__ __forceinline__ v8f wmma_bf16(v16bf a, v16bf b, v8f c) {
  return __builtin_amdgcn_wmma_f32_16x16x32_bf16(false, a, false, b, (short)0, c,
                                                 false, false);
}

// ---------------- conversion kernels ----------------

__global__ void cvt_bf16_kernel(const float* __restrict__ src,
                                bf16_t* __restrict__ dst, int n) {
  int i = blockIdx.x * blockDim.x + threadIdx.x;
  int stride = gridDim.x * blockDim.x;
  for (; i < n; i += stride) dst[i] = (bf16_t)src[i];
}

// Wt[widx][n][c] = W_widx[c][n]  (bf16, so B fragments load contiguously)
__global__ void cvt_w_kernel(const float* __restrict__ Wq,
                             const float* __restrict__ Wk,
                             const float* __restrict__ Wv,
                             bf16_t* __restrict__ Wt) {
  int idx = blockIdx.x * blockDim.x + threadIdx.x;
  int total = 3 * HS_ * C_;
  int stride = gridDim.x * blockDim.x;
  for (; idx < total; idx += stride) {
    int widx = idx / (HS_ * C_);
    int rem  = idx % (HS_ * C_);
    int n = rem / C_, c = rem % C_;
    const float* W = (widx == 0) ? Wq : ((widx == 1) ? Wk : Wv);
    Wt[idx] = (bf16_t)W[c * HS_ + n];
  }
}

// E -> bf16, zero-padded to EPAD rows
__global__ void cvt_e_kernel(const float* __restrict__ E,
                             bf16_t* __restrict__ Eb) {
  int idx = blockIdx.x * blockDim.x + threadIdx.x;
  int total = EPAD * HS_;
  int stride = gridDim.x * blockDim.x;
  for (; idx < total; idx += stride)
    Eb[idx] = (idx < NREL * HS_) ? (bf16_t)E[idx] : (bf16_t)0.0f;
}

// ---------------- QKV projection GEMM ----------------
// grid = (M/128, 3); 8 waves/block, each wave: 16 rows x 64 cols, K=768.
// q,k stored [B*T, HS] bf16 row-major; v stored transposed [B][HS][T] bf16.
__global__ __launch_bounds__(256) void qkv_kernel(
    const bf16_t* __restrict__ xb, const bf16_t* __restrict__ Wt,
    bf16_t* __restrict__ qb, bf16_t* __restrict__ kb,
    bf16_t* __restrict__ vTb) {
  int w = threadIdx.x >> 5;
  int lane = threadIdx.x & 31;
  int half = lane >> 4, ln = lane & 15;
  int widx = blockIdx.y;
  int row0 = blockIdx.x * 128 + w * 16;

  const bf16_t* arow = xb + (size_t)(row0 + ln) * C_;
  const bf16_t* Wsel = Wt + (size_t)widx * HS_ * C_;

  v8f acc[4];
  v8f zero = {0.f, 0.f, 0.f, 0.f, 0.f, 0.f, 0.f, 0.f};
#pragma unroll
  for (int nb = 0; nb < 4; ++nb) acc[nb] = zero;

  for (int kk = 0; kk < C_; kk += 32) {
    v16bf a = frag_ld(arow, kk, half);
#pragma unroll
    for (int nb = 0; nb < 4; ++nb) {
      const bf16_t* brow = Wsel + (size_t)(nb * 16 + ln) * C_;
      v16bf bfr = frag_ld(brow, kk, half);
      acc[nb] = wmma_bf16(a, bfr, acc[nb]);
    }
  }

#pragma unroll
  for (int nb = 0; nb < 4; ++nb)
#pragma unroll
    for (int p = 0; p < 8; ++p) {
      int r = row0 + p + 8 * half;        // C-layout row
      int cidx = nb * 16 + ln;            // C-layout col
      float v = acc[nb][p];
      if (widx == 0)      qb[(size_t)r * HS_ + cidx] = (bf16_t)v;
      else if (widx == 1) kb[(size_t)r * HS_ + cidx] = (bf16_t)v;
      else {
        int bB = r >> 10, t = r & (T_ - 1);
        vTb[((size_t)bB * HS_ + cidx) * T_ + t] = (bf16_t)v;
      }
    }
}

// ---------------- flash attention with relative bias ----------------
// 4 waves/block; wave owns one 16-query tile; 32 keys per iteration.
__global__ __launch_bounds__(128) void fa_rel_kernel(
    const bf16_t* __restrict__ qb, const bf16_t* __restrict__ kb,
    const bf16_t* __restrict__ vTb, const bf16_t* __restrict__ Eb,
    float* __restrict__ out) {
  __shared__ __align__(16) bf16_t pbuf[4][16][32];  // P staging, per wave

  int w = threadIdx.x >> 5;
  int lane = threadIdx.x & 31;
  int half = lane >> 4, ln = lane & 15;

  int tile = blockIdx.x * 4 + w;          // 0..1023
  int b  = tile >> 6;
  int t0 = (tile & 63) << 4;

  // Persistent Q A-fragments (16 x 64 = two k-steps)
  const bf16_t* qrow = qb + ((size_t)(b << 10) + t0 + ln) * HS_;
  v16bf qa0 = frag_ld(qrow, 0, half);
  v16bf qa1 = frag_ld(qrow, 32, half);

  v8f zero = {0.f, 0.f, 0.f, 0.f, 0.f, 0.f, 0.f, 0.f};
  v8f o[4];
#pragma unroll
  for (int nb = 0; nb < 4; ++nb) o[nb] = zero;
  float m[8], lsum[8];
#pragma unroll
  for (int p = 0; p < 8; ++p) { m[p] = -1e30f; lsum[p] = 0.0f; }

  for (int u0 = 0; u0 < T_; u0 += 32) {
    // ---- content scores: two 16x16 tiles, K = HS = 64 ----
    v8f s[2];
#pragma unroll
    for (int tau = 0; tau < 2; ++tau) {
      const bf16_t* krow = kb + ((size_t)(b << 10) + u0 + tau * 16 + ln) * HS_;
      v16bf b0 = frag_ld(krow, 0, half);
      v16bf b1 = frag_ld(krow, 32, half);
      v8f a = zero;
      a = wmma_bf16(qa0, b0, a);
      a = wmma_bf16(qa1, b1, a);
      s[tau] = a;
    }

    // ---- relative window GEMM: G = Qtile @ E[rbase .. rbase+47]^T ----
    int rbase = u0 - t0 + 1008;           // always in [0, 2047]; E padded
    v8f G[3];
#pragma unroll
    for (int g = 0; g < 3; ++g) {
      const bf16_t* erow = Eb + (size_t)(rbase + g * 16 + ln) * HS_;
      v16bf b0 = frag_ld(erow, 0, half);
      v16bf b1 = frag_ld(erow, 32, half);
      v8f a = zero;
      a = wmma_bf16(qa0, b0, a);
      a = wmma_bf16(qa1, b1, a);
      G[g] = a;
    }

    // ---- skew fix-up + online softmax, row by row (C-layout rows) ----
#pragma unroll
    for (int p = 0; p < 8; ++p) {
      int i = p + 8 * half;               // query row within tile
      float sv[2];
#pragma unroll
      for (int tau = 0; tau < 2; ++tau) {
        int j2 = tau * 16 + ln;           // key col within 32-key step
        int c = j2 - i + 15;              // 0..46 window column
        int sl = (c & 15) + 16 * half;    // same VGPR p, shifted lane
        float g0 = __shfl(G[0][p], sl, 32);
        float g1 = __shfl(G[1][p], sl, 32);
        float g2 = __shfl(G[2][p], sl, 32);
        float rel = (c < 16) ? g0 : ((c < 32) ? g1 : g2);
        sv[tau] = (s[tau][p] + rel) * 0.125f;  // HS^-0.5
      }
      // row-max over 32 cols (reduce within 16-lane half)
      float cm = fmaxf(sv[0], sv[1]);
#pragma unroll
      for (int d = 8; d >= 1; d >>= 1) cm = fmaxf(cm, __shfl_xor(cm, d, 32));
      float mnew = fmaxf(m[p], cm);
      float alpha = __expf(m[p] - mnew);
      float e0 = __expf(sv[0] - mnew);
      float e1 = __expf(sv[1] - mnew);
      float ps = e0 + e1;
#pragma unroll
      for (int d = 8; d >= 1; d >>= 1) ps += __shfl_xor(ps, d, 32);
      lsum[p] = lsum[p] * alpha + ps;
      m[p] = mnew;
#pragma unroll
      for (int nb = 0; nb < 4; ++nb) o[nb][p] *= alpha;
      pbuf[w][i][ln]      = (bf16_t)e0;   // stage P for A-layout reload
      pbuf[w][i][16 + ln] = (bf16_t)e1;
    }

    // ---- P @ V over 32 keys (A via LDS, B contiguous from vT) ----
    v16bf pa = frag_ld(&pbuf[w][ln][0], 0, half);
#pragma unroll
    for (int nb = 0; nb < 4; ++nb) {
      const bf16_t* vrow =
          vTb + ((size_t)b * HS_ + nb * 16 + ln) * T_ + u0;
      v16bf vb = frag_ld(vrow, 0, half);
      o[nb] = wmma_bf16(pa, vb, o[nb]);
    }
  }

  // ---- normalize and write fp32 output ----
#pragma unroll
  for (int p = 0; p < 8; ++p) {
    float inv = 1.0f / lsum[p];
    int r = (b << 10) + t0 + p + 8 * half;
#pragma unroll
    for (int nb = 0; nb < 4; ++nb)
      out[(size_t)r * HS_ + nb * 16 + ln] = o[nb][p] * inv;
  }
}

// ---------------- host launcher ----------------

extern "C" void kernel_launch(void* const* d_in, const int* in_sizes, int n_in,
                              void* d_out, int out_size, void* d_ws,
                              size_t ws_size, hipStream_t stream) {
  const float* x  = (const float*)d_in[0];
  const float* Wq = (const float*)d_in[1];
  const float* Wk = (const float*)d_in[2];
  const float* Wv = (const float*)d_in[3];
  const float* E  = (const float*)d_in[4];
  float* out = (float*)d_out;

  // workspace carve-up (256B aligned)
  char* ws = (char*)d_ws;
  size_t off = 0;
  auto take = [&](size_t bytes) {
    char* p = ws + off;
    off += (bytes + 255) & ~(size_t)255;
    return p;
  };
  bf16_t* xb  = (bf16_t*)take((size_t)B_ * T_ * C_ * sizeof(bf16_t));
  bf16_t* Wt  = (bf16_t*)take((size_t)3 * HS_ * C_ * sizeof(bf16_t));
  bf16_t* Eb  = (bf16_t*)take((size_t)EPAD * HS_ * sizeof(bf16_t));
  bf16_t* qb  = (bf16_t*)take((size_t)B_ * T_ * HS_ * sizeof(bf16_t));
  bf16_t* kb  = (bf16_t*)take((size_t)B_ * T_ * HS_ * sizeof(bf16_t));
  bf16_t* vTb = (bf16_t*)take((size_t)B_ * HS_ * T_ * sizeof(bf16_t));
  (void)ws_size; (void)in_sizes; (void)n_in; (void)out_size;

  // 1) convert inputs to bf16 layouts
  int nx = B_ * T_ * C_;
  cvt_bf16_kernel<<<4096, 256, 0, stream>>>(x, xb, nx);
  cvt_w_kernel<<<((3 * HS_ * C_) + 255) / 256, 256, 0, stream>>>(Wq, Wk, Wv, Wt);
  cvt_e_kernel<<<((EPAD * HS_) + 255) / 256, 256, 0, stream>>>(E, Eb);

  // 2) QKV projections: 16384 rows / 128 per block, y = which weight
  qkv_kernel<<<dim3((B_ * T_) / 128, 3), 256, 0, stream>>>(xb, Wt, qb, kb, vTb);

  // 3) flash attention with relative bias: 1024 query tiles, 4 per block
  fa_rel_kernel<<<(B_ * (T_ / 16)) / 4, 128, 0, stream>>>(qb, kb, vTb, Eb, out);
}
